// MultiHeadAttention_19018115186711
// MI455X (gfx1250) — matrix operand order repaired and checked by reading, not compiled, not run on hardware
//
#include <hip/hip_runtime.h>
#include <hip/hip_bf16.h>
#include <math.h>

#define DIM     768
#define NHEADS  16
#define HDIM    48
#define HDIMP   64     // padded head dim (pad zeroed via memset)
#define BATCH   4
#define SEQ     2048
#define NTOK    (BATCH*SEQ)

typedef __bf16 bf16_t;
typedef __attribute__((ext_vector_type(16))) __bf16 bf16x16;
typedef __attribute__((ext_vector_type(8)))  __bf16 bf16x8;
typedef __attribute__((ext_vector_type(4)))  __bf16 bf16x4;
typedef __attribute__((ext_vector_type(8)))  float  f32x8;

union FragAB { bf16x16 v; bf16x8 h[2]; };
union Pack8  { bf16x8 v; int i[4]; };

static __device__ inline f32x8 wmma_bf16(const FragAB& a, const FragAB& b, f32x8 c) {
  // D = A(16x32) x B(32x16) + C, f32 accumulate
  return __builtin_amdgcn_wmma_f32_16x16x32_bf16(false, a.v, false, b.v,
                                                 (short)0, c, false, false);
}

// ---------------------------------------------------------------------------
// GEMM:  out[m,n] = sum_k A[m,k] * W[n,k] + bias[n]   (A @ W^T + b)
// MODE 0: A = x (f32); three weights picked by blockIdx.z; bf16 out into
//         Q/K/V buffers [B,H,S,HDIMP]; Q pre-scaled by 1/sqrt(48).
// MODE 1: A = attention output (bf16 [NTOK,DIM]); f32 out to d_out.
// ---------------------------------------------------------------------------
#define GSTR 40

template<int MODE>
__global__ __launch_bounds__(256)
void gemm_wt_bias(const void* __restrict__ Ap,
                  const float* __restrict__ W0, const float* __restrict__ b0,
                  const float* __restrict__ W1, const float* __restrict__ b1,
                  const float* __restrict__ W2, const float* __restrict__ b2,
                  bf16_t* __restrict__ Qb, bf16_t* __restrict__ Kb,
                  bf16_t* __restrict__ Vb, float* __restrict__ Out)
{
  const int tid  = threadIdx.x;
  const int wave = tid >> 5, lane = tid & 31, hh = lane >> 4, lm = lane & 15;
  const int mBase = blockIdx.x * 64;
  const int nBase = blockIdx.y * 64;

  const float* W; const float* bias; bf16_t* dst = nullptr; float pscale = 1.0f;
  if constexpr (MODE == 0) {
    const int z = blockIdx.z;
    W    = (z == 0) ? W0 : (z == 1) ? W1 : W2;
    bias = (z == 0) ? b0 : (z == 1) ? b1 : b2;
    dst  = (z == 0) ? Qb : (z == 1) ? Kb : Vb;
    if (z == 0) pscale = 0.14433756729740643f;  // 1/sqrt(48) folded into Q
  } else {
    W = W0; bias = b0;
  }

  __shared__ __align__(16) bf16_t Als[64 * GSTR];   // A tile [m][k]
  __shared__ __align__(16) bf16_t Bls[64 * GSTR];   // W tile [n][k]

  const int tr  = wave & 3;
  const int tcp = wave >> 2;

  f32x8 acc0 = {}; f32x8 acc1 = {};

  for (int kb = 0; kb < DIM; kb += 32) {
    __syncthreads();
    for (int i = tid; i < 64 * 8; i += 256) {
      const int r = i >> 3, c4 = (i & 7) * 4;
      if constexpr (MODE == 0) {
        const float* src = (const float*)Ap;
        float4 v = *(const float4*)&src[(size_t)(mBase + r) * DIM + kb + c4];
        Als[r*GSTR + c4 + 0] = (bf16_t)v.x;
        Als[r*GSTR + c4 + 1] = (bf16_t)v.y;
        Als[r*GSTR + c4 + 2] = (bf16_t)v.z;
        Als[r*GSTR + c4 + 3] = (bf16_t)v.w;
      } else {
        const bf16_t* src = (const bf16_t*)Ap;
        *(bf16x4*)&Als[r*GSTR + c4] =
            *(const bf16x4*)&src[(size_t)(mBase + r) * DIM + kb + c4];
      }
      float4 w = *(const float4*)&W[(size_t)(nBase + r) * DIM + kb + c4];
      Bls[r*GSTR + c4 + 0] = (bf16_t)w.x;
      Bls[r*GSTR + c4 + 1] = (bf16_t)w.y;
      Bls[r*GSTR + c4 + 2] = (bf16_t)w.z;
      Bls[r*GSTR + c4 + 3] = (bf16_t)w.w;
    }
    __syncthreads();

    // Issue all fragment loads up front, then both WMMAs.
    FragAB a, bt0, bt1;
    a.h[0]   = *(const bf16x8*)&Als[(16*tr + lm)*GSTR + 16*hh];
    a.h[1]   = *(const bf16x8*)&Als[(16*tr + lm)*GSTR + 16*hh + 8];
    const int col0 = 32*tcp + lm;
    const int col1 = 32*tcp + 16 + lm;
    bt0.h[0] = *(const bf16x8*)&Bls[col0*GSTR + 16*hh];
    bt0.h[1] = *(const bf16x8*)&Bls[col0*GSTR + 16*hh + 8];
    bt1.h[0] = *(const bf16x8*)&Bls[col1*GSTR + 16*hh];
    bt1.h[1] = *(const bf16x8*)&Bls[col1*GSTR + 16*hh + 8];

    acc0 = wmma_bf16(a, bt0, acc0);
    acc1 = wmma_bf16(a, bt1, acc1);
  }

  #pragma unroll
  for (int t = 0; t < 2; ++t) {
    const f32x8 acc = t ? acc1 : acc0;
    const int n = nBase + 32*tcp + 16*t + lm;
    const float bv = bias[n];
    #pragma unroll
    for (int r = 0; r < 8; ++r) {
      const int m = mBase + 16*tr + r + 8*hh;
      const float val = (acc[r] + bv) * pscale;
      if constexpr (MODE == 0) {
        const int b = m >> 11, s = m & (SEQ - 1);
        const int hd = n / HDIM, dh = n % HDIM;
        dst[((size_t)(b*NHEADS + hd) * SEQ + s) * HDIMP + dh] = (bf16_t)val;
      } else {
        Out[(size_t)m * DIM + n] = val;
      }
    }
  }
}

// ---------------------------------------------------------------------------
// Flash attention (transposed): S^T = K*Q^T, O^T = V^T*P^T, queries on lanes.
// K tile staged via CDNA5 async global->LDS DMA (ASYNCcnt); V staged with an
// explicit transpose. All fragment loads issued ahead of their consumers so
// LDS latency overlaps softmax VALU work.
// ---------------------------------------------------------------------------
#define KSTR 72
#define VSTR 40

__global__ __launch_bounds__(256)
void attn_kernel(const bf16_t* __restrict__ Q, const bf16_t* __restrict__ K,
                 const bf16_t* __restrict__ V, bf16_t* __restrict__ AO)
{
  const int tid  = threadIdx.x;
  const int wave = tid >> 5, lane = tid & 31, hh = lane >> 4, lm = lane & 15;
  const int bh = blockIdx.y;
  const int b = bh >> 4, h = bh & 15;
  const size_t base = (size_t)bh * SEQ * HDIMP;
  const int q0 = blockIdx.x * 128 + wave * 16;

  __shared__ __align__(16) bf16_t Kls[32 * KSTR];   // K tile [key][dh]
  __shared__ __align__(16) bf16_t Vt [64 * VSTR];   // V tile transposed [dh][key]

  // Q^T fragments (B layout: lane = query column, 16 contiguous dh at 16*hh)
  FragAB qb[2];
  {
    const bf16_t* qrow = Q + base + (size_t)(q0 + lm) * HDIMP;
    #pragma unroll
    for (int c = 0; c < 2; ++c) {
      qb[c].h[0] = *(const bf16x8*)&qrow[c*32 + 16*hh];
      qb[c].h[1] = *(const bf16x8*)&qrow[c*32 + 16*hh + 8];
    }
  }

  f32x8 Ov[4] = {};                 // O^T: 64 dh rows x 16 query cols
  float mcur = -1e30f, lcur = 0.0f; // per-lane (per-query) softmax state

  const int sky = tid >> 3;         // staging: key row (0..31)
  const int sch = tid & 7;          // staging: 8-elt dh chunk

  const unsigned kls_off =
      (unsigned)(size_t)&Kls[sky * KSTR + sch * 8];   // LDS byte offset

  for (int j = 0; j < SEQ / 32; ++j) {
    const int key0 = j * 32;
    __syncthreads();

    // --- K tile: async global->LDS (one b128 DMA per lane, ASYNCcnt). ---
    {
      const bf16_t* gsrc = K + base + (size_t)(key0 + sky) * HDIMP + sch * 8;
      unsigned long long ga = (unsigned long long)(size_t)gsrc;
      unsigned ldst = kls_off;
      asm volatile("global_load_async_to_lds_b128 %0, %1, off"
                   :: "v"(ldst), "v"(ga) : "memory");
    }
    // --- V tile: b128 global read, transposed scalar LDS writes. ---
    {
      bf16x8 vv = *(const bf16x8*)&V[base + (size_t)(key0 + sky) * HDIMP + sch * 8];
      #pragma unroll
      for (int jj = 0; jj < 8; ++jj)
        Vt[(sch*8 + jj) * VSTR + sky] = vv[jj];
    }
    if (j + 1 < SEQ / 32) {   // global_prefetch_b8 next tiles
      __builtin_prefetch(K + base + (size_t)(key0 + 32) * HDIMP + tid * 8, 0, 1);
      __builtin_prefetch(V + base + (size_t)(key0 + 32) * HDIMP + tid * 8, 0, 1);
    }
    asm volatile("s_wait_asynccnt 0x0" ::: "memory");  // own DMA done
    __syncthreads();                                   // publish to block

    // --- Issue ALL K fragment loads, then the 4 score WMMAs. ---
    FragAB kf[2][2];
    #pragma unroll
    for (int sub = 0; sub < 2; ++sub) {
      const int krow = sub*16 + lm;
      kf[sub][0].h[0] = *(const bf16x8*)&Kls[krow*KSTR + 16*hh];
      kf[sub][0].h[1] = *(const bf16x8*)&Kls[krow*KSTR + 16*hh + 8];
      kf[sub][1].h[0] = *(const bf16x8*)&Kls[krow*KSTR + 32 + 16*hh];
      kf[sub][1].h[1] = *(const bf16x8*)&Kls[krow*KSTR + 32 + 16*hh + 8];
    }
    f32x8 sc[2];
    #pragma unroll
    for (int sub = 0; sub < 2; ++sub) {
      f32x8 a = {};
      a = wmma_bf16(kf[sub][0], qb[0], a);
      a = wmma_bf16(kf[sub][1], qb[1], a);
      sc[sub] = a;   // rows: keys {sub*16 + 8*hh + r}, col: query lm
    }

    // --- Issue V fragment loads now; they are consumed after softmax. ---
    FragAB vf[4];
    #pragma unroll
    for (int t = 0; t < 4; ++t) {
      const int dh = t*16 + lm;
      vf[t].h[0] = *(const bf16x8*)&Vt[dh*VSTR + 16*hh];
      vf[t].h[1] = *(const bf16x8*)&Vt[dh*VSTR + 16*hh + 8];
    }

    // --- Per-query softmax: 16 local values + one xor-16 shuffle. ---
    float mx = -1e30f;
    #pragma unroll
    for (int r = 0; r < 8; ++r) mx = fmaxf(mx, fmaxf(sc[0][r], sc[1][r]));
    mx = fmaxf(mx, __shfl_xor(mx, 16, 32));
    const float mnew  = fmaxf(mcur, mx);
    const float alpha = __expf(mcur - mnew);
    mcur = mnew;

    Pack8 p0, p1;
    float rs = 0.0f;
    #pragma unroll
    for (int r = 0; r < 8; ++r) {
      const float e0 = __expf(sc[0][r] - mnew);
      const float e1 = __expf(sc[1][r] - mnew);
      rs += e0 + e1;
      p0.v[r] = (bf16_t)e0;
      p1.v[r] = (bf16_t)e1;
    }
    rs += __shfl_xor(rs, 16, 32);
    lcur = lcur * alpha + rs;

    #pragma unroll
    for (int t = 0; t < 4; ++t)
      #pragma unroll
      for (int r = 0; r < 8; ++r)
        Ov[t][r] *= alpha;

    // --- P^T B-fragment via 4-dword partner exchange (lane ^ 16). ---
    Pack8 send = hh ? p0 : p1;
    Pack8 recv;
    #pragma unroll
    for (int kk = 0; kk < 4; ++kk) recv.i[kk] = __shfl_xor(send.i[kk], 16, 32);
    FragAB pf;
    pf.h[0] = hh ? recv.v : p0.v;   // keys 16*hh + 0..7
    pf.h[1] = hh ? p1.v   : recv.v; // keys 16*hh + 8..15

    // --- O^T += V^T @ P^T ---
    #pragma unroll
    for (int t = 0; t < 4; ++t)
      Ov[t] = wmma_bf16(vf[t], pf, Ov[t]);
  }

  // Finalize: per-lane scalar 1/l, contiguous b128 stores (chunk 3 = pad).
  const float inv = 1.0f / lcur;
  const size_t tok = (size_t)b * SEQ + q0 + lm;
  #pragma unroll
  for (int t = 0; t < 3; ++t) {
    bf16x8 outv;
    #pragma unroll
    for (int r = 0; r < 8; ++r) outv[r] = (bf16_t)(Ov[t][r] * inv);
    *(bf16x8*)&AO[tok * DIM + h*HDIM + t*16 + 8*hh] = outv;
  }
}

// ---------------------------------------------------------------------------
extern "C" void kernel_launch(void* const* d_in, const int* in_sizes, int n_in,
                              void* d_out, int out_size, void* d_ws, size_t ws_size,
                              hipStream_t stream)
{
  const float* x  = (const float*)d_in[0];
  const float* Wq = (const float*)d_in[1];
  const float* bq = (const float*)d_in[2];
  const float* Wk = (const float*)d_in[3];
  const float* bk = (const float*)d_in[4];
  const float* Wv = (const float*)d_in[5];
  const float* bv = (const float*)d_in[6];
  const float* Wo = (const float*)d_in[7];
  const float* bo = (const float*)d_in[8];
  float* out = (float*)d_out;

  const size_t headbuf = (size_t)BATCH * NHEADS * SEQ * HDIMP * sizeof(bf16_t); // 16 MB
  char* ws = (char*)d_ws;
  bf16_t* Qb = (bf16_t*)(ws);
  bf16_t* Kb = (bf16_t*)(ws + headbuf);
  bf16_t* Vb = (bf16_t*)(ws + 2 * headbuf);
  bf16_t* AO = (bf16_t*)(ws + 3 * headbuf);

  // Zero Q/K/V so the dh 48..63 pad contributes nothing.
  hipMemsetAsync(ws, 0, 3 * headbuf, stream);

  dim3 g1(NTOK / 64, DIM / 64, 3);
  gemm_wt_bias<0><<<g1, 256, 0, stream>>>(x, Wq, bq, Wk, bk, Wv, bv,
                                          Qb, Kb, Vb, nullptr);

  dim3 g2(SEQ / 128, BATCH * NHEADS);
  attn_kernel<<<g2, 256, 0, stream>>>(Qb, Kb, Vb, AO);

  dim3 g3(NTOK / 64, DIM / 64, 1);
  gemm_wt_bias<1><<<g3, 256, 0, stream>>>(AO, Wo, bo,
                                          nullptr, nullptr, nullptr, nullptr,
                                          nullptr, nullptr, nullptr, out);
}
